// LSTM_29549374996652
// MI455X (gfx1250) — matrix operand order repaired and checked by reading, not compiled, not run on hardware
//
#include <hip/hip_runtime.h>
#include <hip/hip_bf16.h>

// ---------------------------------------------------------------------------
// Shapes
// ---------------------------------------------------------------------------
#define T_STEPS 4096
#define HIDDEN  256
#define NUM_OUT 18
#define FEAT    1152            // 6*6*32
#define GATES   1024            // 4*HIDDEN

#define LDA 40                  // LDS leading dim in halves: 80 B, 16B-aligned rows

typedef __attribute__((ext_vector_type(16))) _Float16 v16h;
typedef __attribute__((ext_vector_type(8)))  _Float16 v8h;
typedef __attribute__((ext_vector_type(8)))  float    v8f;

// ---------------------------------------------------------------------------
// WMMA fragment helpers (wave32, ISA 7.12.2 layouts), vectorized LDS access.
//
// A 16x32 f16 (MxK) row-major in LDS, stride LDA:
//   lanes 0-15 -> M=lane, K={0..7,16..23}; lanes 16-31 -> M=lane-16,
//   K={8..15,24..31}.  Each half is a contiguous, 16B-aligned v8h.
// B stored TRANSPOSED in LDS (N-major, stride LDA):
//   lanes 0-15 -> N=lane, K={0..7,16..23}; lanes 16-31 -> N=lane-16, K={8..15,24..31}.
//   Each lane reads two contiguous 8-half runs = 2 x v8h.
// C/D 16x16 f32: VGPR j: lanes 0-15 -> (M=j, N=lane);
//                        lanes 16-31 -> (M=8+j, N=lane-16).
// ---------------------------------------------------------------------------
__device__ __forceinline__ v16h frag_load_a(const _Float16* As) {
  const int lane = threadIdx.x & 31;
  const int m    = lane & 15;
  const int kb   = (lane >> 4) << 3;                   // 0 or 8
  const v8h lo = *(const v8h*)(As + m * LDA + kb);      // K kb..kb+7
  const v8h hi = *(const v8h*)(As + m * LDA + 16 + kb); // K 16+kb..16+kb+7
  return __builtin_shufflevector(lo, hi, 0, 1, 2, 3, 4, 5, 6, 7,
                                 8, 9, 10, 11, 12, 13, 14, 15);
}

__device__ __forceinline__ v16h frag_load_b(const _Float16* Bt, int ncol) {
  const int lane = threadIdx.x & 31;
  const int n    = (lane & 15) + ncol;
  const int kb   = (lane >> 4) << 3;                   // 0 or 8
  const v8h lo = *(const v8h*)(Bt + n * LDA + kb);
  const v8h hi = *(const v8h*)(Bt + n * LDA + 16 + kb);
  return __builtin_shufflevector(lo, hi, 0, 1, 2, 3, 4, 5, 6, 7,
                                 8, 9, 10, 11, 12, 13, 14, 15);
}

__device__ __forceinline__ float elu1(float v) {
  return v > 0.f ? v : (expf(v) - 1.f);
}
__device__ __forceinline__ float sigmoidf(float v) {
  return 1.f / (1.f + expf(-v));
}

// ---------------------------------------------------------------------------
// Implicit-GEMM conv 3x3 stride2 SAME, cout=32, + bias + ELU, f16 NHWC out.
// Grid: (ceil(Hout*Wout/16), T). Block: 64 threads = 2 waves.
// Wave w computes output-channel tile [16w, 16w+16).  K = 9*CIN (zero padded
// to multiple of 32).
// ---------------------------------------------------------------------------
template <int CIN, typename Tin>
__global__ void conv_wmma(const Tin* __restrict__ in, const float* __restrict__ w,
                          const float* __restrict__ bias, _Float16* __restrict__ out,
                          int Hin, int Win, int Hout, int Wout, int pad) {
  constexpr int K9     = 9 * CIN;
  constexpr int KSTEPS = (K9 + 31) / 32;

  __shared__ __align__(16) _Float16 As[16 * LDA];   // 16 x 32 (M x K)
  __shared__ __align__(16) _Float16 Bt[32 * LDA];   // 32 x 32 (N x K), transposed

  const int img  = blockIdx.y;
  const int m0   = blockIdx.x * 16;
  const int tid  = threadIdx.x;
  const int wave = tid >> 5;
  const int lane = tid & 31;
  const int HW   = Hout * Wout;

  v8f acc = {};

  for (int ks = 0; ks < KSTEPS; ++ks) {
    const int k0 = ks * 32;
    // Stage A: 16 x 32 im2col tile (zero-padded at K and M edges)
    for (int idx = tid; idx < 512; idx += 64) {
      const int r  = idx >> 5;
      const int kk = idx & 31;
      const int kg = k0 + kk;
      const int m  = m0 + r;
      _Float16 v = (_Float16)0.f;
      if (kg < K9 && m < HW) {
        const int c  = kg % CIN;
        const int kw = (kg / CIN) % 3;
        const int kh = kg / (3 * CIN);
        const int oh = m / Wout, ow = m % Wout;
        const int ih = oh * 2 + kh - pad;
        const int iw = ow * 2 + kw - pad;
        if (ih >= 0 && ih < Hin && iw >= 0 && iw < Win)
          v = (_Float16)(float)in[(((long)img * Hin + ih) * Win + iw) * CIN + c];
      }
      As[r * LDA + kk] = v;
    }
    // Stage B transposed: Bt[n][k] = w[(k0+k)*32 + n], f32 -> f16
    for (int idx = tid; idx < 1024; idx += 64) {
      const int r  = idx >> 5;     // k within step
      const int n  = idx & 31;     // output channel
      const int kg = k0 + r;
      Bt[n * LDA + r] = (kg < K9) ? (_Float16)w[kg * 32 + n] : (_Float16)0.f;
    }
    __syncthreads();
    v16h a = frag_load_a(As);
    v16h b = frag_load_b(Bt, wave * 16);
    acc = __builtin_amdgcn_wmma_f32_16x16x32_f16(false, a, false, b,
                                                 (short)0, acc, false, false);
    __syncthreads();
  }

  const int n  = wave * 16 + (lane & 15);
  const int mb = (lane >> 4) * 8;
#pragma unroll
  for (int j = 0; j < 8; ++j) {
    const int m = m0 + mb + j;
    if (m < HW)
      out[(((long)img * HW) + m) * 32 + n] = (_Float16)elu1(acc[j] + bias[n]);
  }
}

// ---------------------------------------------------------------------------
// x_gates = feats(4096x1152,f16) @ Wx(1152x1024) + lstm_b  -> f32
// Grid: (4096/16, 1024/128). Block: 256 threads = 8 waves; wave w owns
// N-tile [n0+16w, n0+16w+16).
// A operand staged with GLOBAL_LOAD_ASYNC_TO_LDS_B128 (ASYNCcnt path):
// 16 rows x 4 x 16B aligned chunks = 64 async copies issued by threads 0..63.
// ---------------------------------------------------------------------------
__global__ void xgates_wmma(const _Float16* __restrict__ feats,
                            const float* __restrict__ lstmW,
                            const float* __restrict__ lstmb,
                            float* __restrict__ xg) {
  __shared__ __align__(16) _Float16 As[16 * LDA];    // 16 x 32 (M x K)
  __shared__ __align__(16) _Float16 Bt[128 * LDA];   // 128 x 32 (N x K), transposed

  const int tid  = threadIdx.x;
  const int wave = tid >> 5;
  const int lane = tid & 31;
  const int m0   = blockIdx.x * 16;
  const int n0   = blockIdx.y * 128;

  v8f acc = {};

  for (int ks = 0; ks < FEAT / 32; ++ks) {
    const int k0 = ks * 32;
    // Async DMA: feats[m0+r][k0 + 8c .. k0 + 8c + 7] -> As[r][8c], 16B each
    if (tid < 64) {
      const int r = tid >> 2;        // row 0..15
      const int c = tid & 3;         // 16B chunk 0..3
      const unsigned lds_off =
          (unsigned)(unsigned long long)(const void*)(As + r * LDA + c * 8);
      const unsigned long long gaddr = (unsigned long long)(const void*)(
          feats + (long)(m0 + r) * FEAT + k0 + c * 8);
      asm volatile("global_load_async_to_lds_b128 %0, %1, off"
                   :: "v"(lds_off), "v"(gaddr)
                   : "memory");
    }
    // Stage B transposed with f32->f16 conversion (VALU path)
    for (int idx = tid; idx < 32 * 128; idx += 256) {
      const int r = idx >> 7, n = idx & 127;
      Bt[n * LDA + r] = (_Float16)lstmW[(long)(k0 + r) * GATES + n0 + n];
    }
    asm volatile("s_wait_asynccnt 0x0" ::: "memory");
    __syncthreads();
    v16h a = frag_load_a(As);
    v16h b = frag_load_b(Bt, wave * 16);
    acc = __builtin_amdgcn_wmma_f32_16x16x32_f16(false, a, false, b,
                                                 (short)0, acc, false, false);
    __syncthreads();
  }

  const int n  = n0 + wave * 16 + (lane & 15);
  const int mb = (lane >> 4) * 8;
#pragma unroll
  for (int j = 0; j < 8; ++j)
    xg[(long)(m0 + mb + j) * GATES + n] = acc[j] + lstmb[n];
}

// ---------------------------------------------------------------------------
// Sequential LSTM scan: one persistent workgroup, 1024 threads (32 waves).
// h, c, gates live in LDS; Wh (1 MB) streamed from L2. fp32 exact recurrence.
// gate columns per jnp.split order: [i | j | f | o].
// ---------------------------------------------------------------------------
__global__ void __launch_bounds__(1024)
lstm_scan(const float* __restrict__ xg, const float* __restrict__ lstmW,
          const float* __restrict__ c0, const float* __restrict__ h0,
          float* __restrict__ hseq) {
  const float* Wh = lstmW + (long)FEAT * GATES;   // rows FEAT..FEAT+HIDDEN-1
  __shared__ float hs[HIDDEN];
  __shared__ float cs[HIDDEN];
  __shared__ float gs[GATES];

  const int tid = threadIdx.x;
  if (tid < HIDDEN) { hs[tid] = h0[tid]; cs[tid] = c0[tid]; }
  __syncthreads();

  for (int t = 0; t < T_STEPS; ++t) {
    // gates[tid] = xg[t][tid] + dot(h, Wh[:, tid])
    float g = xg[(long)t * GATES + tid];
    if (t + 1 < T_STEPS)
      __builtin_prefetch(&xg[(long)(t + 1) * GATES + tid], 0, 1);
#pragma unroll 4
    for (int k = 0; k < HIDDEN; ++k)
      g = fmaf(hs[k], Wh[(long)k * GATES + tid], g);
    gs[tid] = g;
    __syncthreads();                 // gates ready; everyone done reading hs

    if (tid < HIDDEN) {
      const float gi = gs[tid];
      const float gj = gs[HIDDEN + tid];
      const float gf = gs[2 * HIDDEN + tid];
      const float go = gs[3 * HIDDEN + tid];
      const float nc = cs[tid] * sigmoidf(gf + 1.f) + sigmoidf(gi) * tanhf(gj);
      const float nh = tanhf(nc) * sigmoidf(go);
      cs[tid] = nc;
      hs[tid] = nh;
      hseq[(long)t * HIDDEN + tid] = nh;
    }
    __syncthreads();                 // h updated before next step reads it
  }
}

// ---------------------------------------------------------------------------
// logits = h_seq @ act_W + act_b ; one block per timestep.
// ---------------------------------------------------------------------------
__global__ void logits_kernel(const float* __restrict__ hseq,
                              const float* __restrict__ actW,
                              const float* __restrict__ actb,
                              float* __restrict__ logits) {
  __shared__ float hrow[HIDDEN];
  const int t = blockIdx.x;
  for (int k = threadIdx.x; k < HIDDEN; k += 64)
    hrow[k] = hseq[(long)t * HIDDEN + k];
  __syncthreads();
  if (threadIdx.x < NUM_OUT) {
    float s = actb[threadIdx.x];
#pragma unroll 4
    for (int k = 0; k < HIDDEN; ++k)
      s = fmaf(hrow[k], actW[k * NUM_OUT + threadIdx.x], s);
    logits[(long)t * NUM_OUT + threadIdx.x] = s;
  }
}

// ---------------------------------------------------------------------------
// Launch
// ---------------------------------------------------------------------------
extern "C" void kernel_launch(void* const* d_in, const int* in_sizes, int n_in,
                              void* d_out, int out_size, void* d_ws, size_t ws_size,
                              hipStream_t stream) {
  const float* x      = (const float*)d_in[0];
  const float* c_in   = (const float*)d_in[1];
  const float* h_in   = (const float*)d_in[2];
  const float* W1     = (const float*)d_in[3];
  const float* b1     = (const float*)d_in[4];
  const float* W2     = (const float*)d_in[5];
  const float* b2     = (const float*)d_in[6];
  const float* W3     = (const float*)d_in[7];
  const float* b3     = (const float*)d_in[8];
  const float* W4     = (const float*)d_in[9];
  const float* b4     = (const float*)d_in[10];
  const float* lstm_W = (const float*)d_in[11];
  const float* lstm_b = (const float*)d_in[12];
  const float* act_W  = (const float*)d_in[13];
  const float* act_b  = (const float*)d_in[14];

  // Workspace carve-up (f16 conv activations, f32 x_gates)
  const long N1 = (long)T_STEPS * 42 * 42 * 32;   // conv1 out
  const long N2 = (long)T_STEPS * 21 * 21 * 32;   // conv2 out
  const long N3 = (long)T_STEPS * 11 * 11 * 32;   // conv3 out
  const long N4 = (long)T_STEPS * 6 * 6 * 32;     // conv4 out (= feats)
  _Float16* c1    = (_Float16*)d_ws;
  _Float16* c2    = c1 + N1;
  _Float16* c3    = c2 + N2;
  _Float16* feats = c3 + N3;
  float*    xg    = (float*)(feats + N4);

  float* logits = (float*)d_out;
  float* hseq   = logits + (long)T_STEPS * NUM_OUT;

  // Conv stack (SAME, stride 2): pads 84->42:0, 42->21:0, 21->11:1, 11->6:1
  conv_wmma<4, float><<<dim3((42 * 42 + 15) / 16, T_STEPS), 64, 0, stream>>>(
      x, W1, b1, c1, 84, 84, 42, 42, 0);
  conv_wmma<32, _Float16><<<dim3((21 * 21 + 15) / 16, T_STEPS), 64, 0, stream>>>(
      c1, W2, b2, c2, 42, 42, 21, 21, 0);
  conv_wmma<32, _Float16><<<dim3((11 * 11 + 15) / 16, T_STEPS), 64, 0, stream>>>(
      c2, W3, b3, c3, 21, 21, 11, 11, 1);
  conv_wmma<32, _Float16><<<dim3((6 * 6 + 15) / 16, T_STEPS), 64, 0, stream>>>(
      c3, W4, b4, feats, 11, 11, 6, 6, 1);

  // feats @ Wx + b
  xgates_wmma<<<dim3(T_STEPS / 16, GATES / 128), 256, 0, stream>>>(
      feats, lstm_W, lstm_b, xg);

  // Sequential recurrence (persistent single workgroup)
  lstm_scan<<<1, 1024, 0, stream>>>(xg, lstm_W, c_in, h_in, hseq);

  // Action head
  logits_kernel<<<T_STEPS, 64, 0, stream>>>(hseq, act_W, act_b, logits);
}
